// lstm_Model_3186865734078
// MI455X (gfx1250) — matrix operand order repaired and checked by reading, not compiled, not run on hardware
//
#include <hip/hip_runtime.h>
#include <hip/hip_bf16.h>

typedef __attribute__((ext_vector_type(16))) _Float16 v16h;
typedef __attribute__((ext_vector_type(8)))  float    v8f;

#define BSZ 512
#define LSZ 1000
#define HSZ 128
#define GSZ 512   // 4*H

// ---- fast activations (NaN-free tanh form) ----
__device__ __forceinline__ float sigm(float v) { return 1.0f / (1.0f + __expf(-v)); }
__device__ __forceinline__ float tanh2(float v) {
  float e = __expf(2.0f * v);
  return 1.0f - 2.0f / (e + 1.0f);
}

// A-fragment (16x32 f16, MxK) from an f16 row-major [16][128] buffer.
// halves h -> K = k0 + h + 16*(lane>=16)
__device__ __forceinline__ v16h loadAfrag(const _Float16* hbuf, int m, int k0, int l16) {
  union { v16h v; unsigned int u[8]; } f;
  const unsigned int* p1 = (const unsigned int*)(hbuf + m * HSZ + k0 + l16 * 16);
#pragma unroll
  for (int q = 0; q < 8; ++q) f.u[q] = p1[q];
  return f.v;
}

// B-fragment (32x16 f16, KxN) from f16 row-major W[rows][128]; B[k][n] = W[n][k].
// lane holds column n = (lane&15); halves h -> K = k0 + h + 16*(lane>=16)
__device__ __forceinline__ v16h loadBfrag(const _Float16* w, int row, int k0, int l16) {
  union { v16h v; unsigned int u[8]; } f;
  const unsigned int* p = (const unsigned int*)(w + row * HSZ + k0 + l16 * 16);
#pragma unroll
  for (int q = 0; q < 8; ++q) f.u[q] = p[q];
  return f.v;
}

__global__ __launch_bounds__(256, 1)
void lstm_persistent(const float* __restrict__ x,
                     const float* __restrict__ w_ih1, const float* __restrict__ w_hh1,
                     const float* __restrict__ b_ih1, const float* __restrict__ b_hh1,
                     const float* __restrict__ w_ih2, const float* __restrict__ w_hh2,
                     const float* __restrict__ b_ih2, const float* __restrict__ b_hh2,
                     const float* __restrict__ w_lin, const float* __restrict__ b_lin,
                     float* __restrict__ out) {
  __shared__ _Float16 w2a[GSZ * HSZ];       // w_ih2 as f16   (128 KB)
  __shared__ _Float16 w2b[GSZ * HSZ];       // w_hh2 as f16   (128 KB)
  __shared__ _Float16 h1buf[2][16 * HSZ];   // ping-pong layer-1 hidden (2x4 KB)
  __shared__ _Float16 h2buf[2][16 * HSZ];   // ping-pong layer-2 hidden (2x4 KB)
  __shared__ _Float16 wlt[16 * HSZ];        // row 0 = w_lin, rows 1..15 = 0 (4 KB)

  const int tid  = threadIdx.x;
  const int lane = tid & 31;
  const int wv   = tid >> 5;          // wave 0..7
  const int ln   = lane & 15;
  const int l16  = lane >> 4;         // 0 or 1
  const int m0   = blockIdx.x * 16;   // batch tile base

  // ---- one-time preload: layer-2 weights -> LDS f16, head weights, zero h state ----
  for (int i = tid; i < GSZ * HSZ; i += 256) w2a[i] = (_Float16)w_ih2[i];
  for (int i = tid; i < GSZ * HSZ; i += 256) w2b[i] = (_Float16)w_hh2[i];
  for (int i = tid; i < 16 * HSZ; i += 256) {
    wlt[i] = (i < HSZ) ? (_Float16)w_lin[i] : (_Float16)0.0f;
    h1buf[0][i] = (_Float16)0.0f; h1buf[1][i] = (_Float16)0.0f;
    h2buf[0][i] = (_Float16)0.0f; h2buf[1][i] = (_Float16)0.0f;
  }

  // ---- per-wave column ownership: tile g (=gate) covers j = g*128 + wv*16 + ln ----
  int   jr[4];
  float b1v[4], b2v[4], wxv[4];
  v16h  w1f[4][4];                    // W_hh1 B-fragments in registers
#pragma unroll
  for (int g = 0; g < 4; ++g) {
    int j = g * HSZ + wv * 16 + ln;
    jr[g]  = j;
    b1v[g] = b_ih1[j] + b_hh1[j];
    b2v[g] = b_ih2[j] + b_hh2[j];
    wxv[g] = w_ih1[j];                // w_ih1 is [4H,1]
#pragma unroll
    for (int kc = 0; kc < 4; ++kc) {
      const float4* s = (const float4*)(w_hh1 + (size_t)j * HSZ + kc * 32 + l16 * 16);
      union { v16h v; } f;
#pragma unroll
      for (int q = 0; q < 4; ++q) {
        float4 t = s[q];
        f.v[q * 4 + 0] = (_Float16)t.x; f.v[q * 4 + 1] = (_Float16)t.y;
        f.v[q * 4 + 2] = (_Float16)t.z; f.v[q * 4 + 3] = (_Float16)t.w;
      }
      w1f[g][kc] = f.v;
    }
  }

  v8f c1 = {}; v8f c2 = {};
  const float blin = b_lin[0];
  __syncthreads();

  // One LSTM step. Reads h*r (state t-1), writes h*w (state t). Two barriers.
  auto step = [&](int t, const float (&xr)[8],
                  const _Float16* h1r, _Float16* h1w,
                  const _Float16* h2r, _Float16* h2w) {
    // ---------- layer 1: gates1 = x*w_ih1^T + b + h1 @ W_hh1^T ----------
    v16h a1[4];
#pragma unroll
    for (int kc = 0; kc < 4; ++kc) a1[kc] = loadAfrag(h1r, ln, kc * 32, l16);

    v8f acc[4];
#pragma unroll
    for (int g = 0; g < 4; ++g)
#pragma unroll
      for (int r = 0; r < 8; ++r) acc[g][r] = xr[r] * wxv[g] + b1v[g];

#pragma unroll
    for (int g = 0; g < 4; ++g)
#pragma unroll
      for (int kc = 0; kc < 4; ++kc)
        acc[g] = __builtin_amdgcn_wmma_f32_16x16x32_f16(
            false, a1[kc], false, w1f[g][kc], (short)0, acc[g], false, false);

#pragma unroll
    for (int r = 0; r < 8; ++r) {
      float iv = sigm(acc[0][r]);
      float fv = sigm(acc[1][r]);
      float gv = tanh2(acc[2][r]);
      float ov = sigm(acc[3][r]);
      float cn = fv * c1[r] + iv * gv;
      c1[r] = cn;
      h1w[(r + l16 * 8) * HSZ + wv * 16 + ln] = (_Float16)(ov * tanh2(cn));
    }
    __syncthreads();                                   // h1(t) visible

    // ---------- layer 2: gates2 = h1 @ W_ih2^T + h2 @ W_hh2^T + b ----------
    v16h a1n[4], a2n[4];
#pragma unroll
    for (int kc = 0; kc < 4; ++kc) {
      a1n[kc] = loadAfrag(h1w, ln, kc * 32, l16);      // h1(t)
      a2n[kc] = loadAfrag(h2r, ln, kc * 32, l16);      // h2(t-1)
    }
#pragma unroll
    for (int g = 0; g < 4; ++g)
#pragma unroll
      for (int r = 0; r < 8; ++r) acc[g][r] = b2v[g];

#pragma unroll
    for (int g = 0; g < 4; ++g) {
#pragma unroll
      for (int kc = 0; kc < 4; ++kc)
        acc[g] = __builtin_amdgcn_wmma_f32_16x16x32_f16(
            false, a1n[kc], false, loadBfrag(w2a, jr[g], kc * 32, l16),
            (short)0, acc[g], false, false);
#pragma unroll
      for (int kc = 0; kc < 4; ++kc)
        acc[g] = __builtin_amdgcn_wmma_f32_16x16x32_f16(
            false, a2n[kc], false, loadBfrag(w2b, jr[g], kc * 32, l16),
            (short)0, acc[g], false, false);
    }

#pragma unroll
    for (int r = 0; r < 8; ++r) {
      float iv = sigm(acc[0][r]);
      float fv = sigm(acc[1][r]);
      float gv = tanh2(acc[2][r]);
      float ov = sigm(acc[3][r]);
      float cn = fv * c2[r] + iv * gv;
      c2[r] = cn;
      h2w[(r + l16 * 8) * HSZ + wv * 16 + ln] = (_Float16)(ov * tanh2(cn));
    }
    __syncthreads();                                   // h2(t) visible

    // ---------- output head on wave 0: out[:,t] = h2(t) @ w_lin^T + b_lin ----------
    if (wv == 0) {                                     // wave-uniform branch, EXEC all 1s
      v8f ao;
#pragma unroll
      for (int r = 0; r < 8; ++r) ao[r] = blin;
#pragma unroll
      for (int kc = 0; kc < 4; ++kc)
        ao = __builtin_amdgcn_wmma_f32_16x16x32_f16(
            false, loadAfrag(h2w, ln, kc * 32, l16),
            false, loadBfrag(wlt, ln, kc * 32, l16),
            (short)0, ao, false, false);
      if (ln == 0) {                                   // column N=0 holds the result
#pragma unroll
        for (int r = 0; r < 8; ++r)
          out[(size_t)(m0 + r + l16 * 8) * LSZ + t] = ao[r];
      }
    }
  };

  for (int t = 0; t < LSZ; t += 2) {
    float xa[8], xb[8];
#pragma unroll
    for (int r = 0; r < 8; ++r) {
      float2 v = *(const float2*)(x + (size_t)(m0 + r + l16 * 8) * LSZ + t);
      xa[r] = v.x; xb[r] = v.y;
    }
    step(t,     xa, h1buf[0], h1buf[1], h2buf[0], h2buf[1]);
    step(t + 1, xb, h1buf[1], h1buf[0], h2buf[1], h2buf[0]);
  }
}

extern "C" void kernel_launch(void* const* d_in, const int* in_sizes, int n_in,
                              void* d_out, int out_size, void* d_ws, size_t ws_size,
                              hipStream_t stream) {
  (void)in_sizes; (void)n_in; (void)out_size; (void)d_ws; (void)ws_size;
  const float* x     = (const float*)d_in[0];
  const float* w_ih1 = (const float*)d_in[1];
  const float* w_hh1 = (const float*)d_in[2];
  const float* b_ih1 = (const float*)d_in[3];
  const float* b_hh1 = (const float*)d_in[4];
  const float* w_ih2 = (const float*)d_in[5];
  const float* w_hh2 = (const float*)d_in[6];
  const float* b_ih2 = (const float*)d_in[7];
  const float* b_hh2 = (const float*)d_in[8];
  const float* w_lin = (const float*)d_in[9];
  const float* b_lin = (const float*)d_in[10];
  float* out = (float*)d_out;

  dim3 grid(BSZ / 16, 1, 1);     // 32 workgroups, one 16-row batch tile each
  dim3 block(256, 1, 1);         // 8 waves (wave32)
  lstm_persistent<<<grid, block, 0, stream>>>(x, w_ih1, w_hh1, b_ih1, b_hh1,
                                              w_ih2, w_hh2, b_ih2, b_hh2,
                                              w_lin, b_lin, out);
}
